// MaxViT_34471407517854
// MI455X (gfx1250) — compile-verified, matrix operand order checked by reading, not run on hardware
//
#include <hip/hip_runtime.h>
#include <hip/hip_bf16.h>
#include <stdint.h>

// ---------------------------------------------------------------------------
// Types for CDNA5 WMMA (wave32, V_WMMA_F32_16X16X32_BF16)
// ---------------------------------------------------------------------------
typedef __attribute__((ext_vector_type(16))) __bf16        v16bf;
typedef __attribute__((ext_vector_type(2)))  __bf16        bf16x2;
typedef __attribute__((ext_vector_type(8)))  float         v8f;
typedef __attribute__((ext_vector_type(4)))  float         f32x4;
typedef __attribute__((ext_vector_type(2)))  float         f32x2;
typedef __attribute__((ext_vector_type(4)))  unsigned int  u32x4;
typedef __attribute__((ext_vector_type(2)))  unsigned int  u32x2;
typedef int v4i_vs __attribute__((vector_size(16)));   // matches async-LDS builtin param

#if defined(__HIP_DEVICE_COMPILE__) && __has_builtin(__builtin_amdgcn_global_load_async_to_lds_b128)
#define HAVE_ASYNC_LDS 1
#endif

__device__ __forceinline__ unsigned int f2bf_u(float f) {
  unsigned int u = __float_as_uint(f);
  return (u + 0x7FFFu + ((u >> 16) & 1u)) >> 16;   // round-to-nearest-even
}
__device__ __forceinline__ unsigned int pack2bf(float a, float b) {
#if defined(__HIP_DEVICE_COMPILE__)
  // vector fptrunc -> single v_cvt_pk_bf16_f32 on gfx1250
  f32x2 f = {a, b};
  bf16x2 h = __builtin_convertvector(f, bf16x2);
  unsigned int u; __builtin_memcpy(&u, &h, 4);
  return u;
#else
  return f2bf_u(a) | (f2bf_u(b) << 16);
#endif
}

__device__ __forceinline__ float gelu_f(float x) {
  return 0.5f * x * (1.0f + erff(x * 0.70710678118654752f));
}
__device__ __forceinline__ f32x4 gelu4(f32x4 v) {
  f32x4 r;
  r.x = gelu_f(v.x); r.y = gelu_f(v.y); r.z = gelu_f(v.z); r.w = gelu_f(v.w);
  return r;
}

#if defined(HAVE_ASYNC_LDS)
__device__ __forceinline__ void async_cp_b128(const float* g, float* l) {
  v4i_vs* gp = (v4i_vs*)(float*)g;   // strip const, retype (generic AS)
  v4i_vs* lp = (v4i_vs*)l;
  __builtin_amdgcn_global_load_async_to_lds_b128(
      (__attribute__((address_space(1))) v4i_vs*)gp,
      (__attribute__((address_space(3))) v4i_vs*)lp, 0, 0);
}
__device__ __forceinline__ void wait_async0() {
#if __has_builtin(__builtin_amdgcn_s_wait_asynccnt)
  __builtin_amdgcn_s_wait_asynccnt(0);
#else
  asm volatile("s_wait_asynccnt 0x0" ::: "memory");
#endif
}
#endif

// ---------------------------------------------------------------------------
// WMMA GEMM:  C[M,N] = act(A[M,K] @ Wt[N,K]^T + bias)
// f32 in/out, bf16 WMMA internally (accumulate f32).  REQUIRES K % 32 == 0.
// 256 threads = 8 waves; block tile 128x128; wave tile 32x64 (8 wmma / K-step).
// Double-buffered LDS (one barrier per K-step), float4 global staging with
// row-granular guards, ds_load_b128 fragment reads, global_prefetch hints.
// ---------------------------------------------------------------------------
#define BM 128
#define BN 128
#define BK 32

__global__ __launch_bounds__(256) void k_gemm_bf16(
    const float* __restrict__ A, const float* __restrict__ Wt,
    const float* __restrict__ bias, float* __restrict__ C,
    int M, int N, int K, int act)
{
  __shared__ __attribute__((aligned(16))) unsigned short sA[2][BM * BK];
  __shared__ __attribute__((aligned(16))) unsigned short sB[2][BN * BK];

  const int tid  = threadIdx.x;
  const int lane = tid & 31;
  const int wid  = tid >> 5;
  const int m0   = blockIdx.y * BM;
  const int n0   = blockIdx.x * BN;
  const int wm   = (wid & 3) * 32;
  const int wn   = (wid >> 2) * 64;
  const int half = lane >> 4;
  const int ml   = lane & 15;
  const int srow = tid >> 3;
  const int skq  = (tid & 7) * 4;

  v8f acc[2][4];
  const v8f vz = {0.f,0.f,0.f,0.f,0.f,0.f,0.f,0.f};
  #pragma unroll
  for (int mi = 0; mi < 2; ++mi)
    #pragma unroll
    for (int j = 0; j < 4; ++j) acc[mi][j] = vz;

  f32x4 ra[4], rb[4];
  const f32x4 fz = {0.f, 0.f, 0.f, 0.f};

  auto stage_regs = [&](int k0) {
    #pragma unroll
    for (int t = 0; t < 4; ++t) {
      int row = srow + t * 32;
      int gm = m0 + row;
      const float* pA = A + (size_t)gm * K + k0 + skq;
      int gn = n0 + row;
      const float* pB = Wt + (size_t)gn * K + k0 + skq;
      ra[t] = (gm < M) ? *(const f32x4*)pA : fz;
      rb[t] = (gn < N) ? *(const f32x4*)pB : fz;
      if (gm < M) __builtin_prefetch(pA + BK, 0, 1);
      if (gn < N) __builtin_prefetch(pB + BK, 0, 1);
    }
  };
  auto store_lds = [&](int buf) {
    #pragma unroll
    for (int t = 0; t < 4; ++t) {
      int row = srow + t * 32;
      u32x2 pa = { pack2bf(ra[t].x, ra[t].y), pack2bf(ra[t].z, ra[t].w) };
      u32x2 pb = { pack2bf(rb[t].x, rb[t].y), pack2bf(rb[t].z, rb[t].w) };
      *(u32x2*)&sA[buf][row * BK + skq] = pa;
      *(u32x2*)&sB[buf][row * BK + skq] = pb;
    }
  };

  union Frag { v16bf v; u32x4 q[2]; };
  auto ld_frag = [&](const unsigned short* base) -> v16bf {
    Frag f;
    f.q[0] = *(const u32x4*)(base);
    f.q[1] = *(const u32x4*)(base + 16);
    return f.v;
  };

  stage_regs(0);
  store_lds(0);
  __syncthreads();

  int buf = 0;
  for (int k0 = 0; k0 < K; k0 += BK) {
    const bool more = (k0 + BK) < K;
    if (more) stage_regs(k0 + BK);

    v16bf fa0 = ld_frag(&sA[buf][(wm      + ml) * BK + half * 8]);
    v16bf fa1 = ld_frag(&sA[buf][(wm + 16 + ml) * BK + half * 8]);
    #pragma unroll
    for (int j = 0; j < 4; ++j) {
      v16bf fb = ld_frag(&sB[buf][(wn + j * 16 + ml) * BK + half * 8]);
      acc[0][j] = __builtin_amdgcn_wmma_f32_16x16x32_bf16(false, fa0, false, fb,
                                                          (short)0, acc[0][j], false, false);
      acc[1][j] = __builtin_amdgcn_wmma_f32_16x16x32_bf16(false, fa1, false, fb,
                                                          (short)0, acc[1][j], false, false);
    }

    if (more) store_lds(buf ^ 1);
    __syncthreads();
    buf ^= 1;
  }

  const int mbase = half * 8;
  #pragma unroll
  for (int mi = 0; mi < 2; ++mi) {
    #pragma unroll
    for (int r = 0; r < 8; ++r) {
      int gm = m0 + wm + mi * 16 + mbase + r;
      if (gm >= M) continue;
      #pragma unroll
      for (int j = 0; j < 4; ++j) {
        int gn = n0 + wn + j * 16 + ml;
        if (gn >= N) continue;
        float v = acc[mi][j][r];
        if (bias) v += bias[gn];
        if (act == 1) v = gelu_f(v);
        C[(size_t)gm * N + gn] = v;
      }
    }
  }
}

// ---------------------------------------------------------------------------
// Implicit-im2col WMMA conv GEMM (3x3, pad=1, NHWC, Cin % 32 == 0).
// out[B*Ho*Wo, Cout] = in (*) w  with wr repacked as [Cout][tap(9)][Cin].
// K = 9*Cin; each K-tile of 32 stays inside one (kh,kw) tap since Cin%32==0.
// ---------------------------------------------------------------------------
__global__ __launch_bounds__(256) void k_conv_gemm_bf16(
    const float* __restrict__ in, const float* __restrict__ wr,
    const float* __restrict__ bias, float* __restrict__ out,
    int Hin, int Win, int Cin, int Ho, int Wo,
    int stride, int M, int N, int K)
{
  __shared__ __attribute__((aligned(16))) unsigned short sA[2][BM * BK];
  __shared__ __attribute__((aligned(16))) unsigned short sB[2][BN * BK];

  const int tid  = threadIdx.x;
  const int lane = tid & 31;
  const int wid  = tid >> 5;
  const int m0   = blockIdx.y * BM;
  const int n0   = blockIdx.x * BN;
  const int wm   = (wid & 3) * 32;
  const int wn   = (wid >> 2) * 64;
  const int half = lane >> 4;
  const int ml   = lane & 15;
  const int srow = tid >> 3;
  const int skq  = (tid & 7) * 4;

  // hoist output-pixel decode (k-invariant per staging slot)
  int pb[4], pho[4], pwo[4];
  #pragma unroll
  for (int t = 0; t < 4; ++t) {
    int gm = m0 + srow + t * 32;
    int gmc = (gm < M) ? gm : 0;
    pwo[t] = gmc % Wo;
    int tt = gmc / Wo;
    pho[t] = tt % Ho;
    pb[t]  = tt / Ho;
  }

  v8f acc[2][4];
  const v8f vz = {0.f,0.f,0.f,0.f,0.f,0.f,0.f,0.f};
  #pragma unroll
  for (int mi = 0; mi < 2; ++mi)
    #pragma unroll
    for (int j = 0; j < 4; ++j) acc[mi][j] = vz;

  f32x4 ra[4], rb[4];
  const f32x4 fz = {0.f, 0.f, 0.f, 0.f};

  auto stage_regs = [&](int k0) {
    int kk  = k0 + skq;
    int tap = kk / Cin;
    int ci  = kk - tap * Cin;
    int kh  = tap / 3, kw = tap - kh * 3;
    #pragma unroll
    for (int t = 0; t < 4; ++t) {
      int gm = m0 + srow + t * 32;
      int hi = pho[t] * stride - 1 + kh;
      int wi = pwo[t] * stride - 1 + kw;
      bool ok = (gm < M) && ((unsigned)hi < (unsigned)Hin) && ((unsigned)wi < (unsigned)Win);
      const float* pA = in + (((size_t)pb[t] * Hin + hi) * Win + wi) * Cin + ci;
      ra[t] = ok ? *(const f32x4*)pA : fz;
      int gn = n0 + srow + t * 32;
      const float* pB = wr + (size_t)gn * K + kk;
      rb[t] = (gn < N) ? *(const f32x4*)pB : fz;
      if (gn < N) __builtin_prefetch(pB + BK, 0, 1);
    }
  };
  auto store_lds = [&](int buf) {
    #pragma unroll
    for (int t = 0; t < 4; ++t) {
      int row = srow + t * 32;
      u32x2 pa = { pack2bf(ra[t].x, ra[t].y), pack2bf(ra[t].z, ra[t].w) };
      u32x2 pb2 = { pack2bf(rb[t].x, rb[t].y), pack2bf(rb[t].z, rb[t].w) };
      *(u32x2*)&sA[buf][row * BK + skq] = pa;
      *(u32x2*)&sB[buf][row * BK + skq] = pb2;
    }
  };

  union Frag { v16bf v; u32x4 q[2]; };
  auto ld_frag = [&](const unsigned short* base) -> v16bf {
    Frag f;
    f.q[0] = *(const u32x4*)(base);
    f.q[1] = *(const u32x4*)(base + 16);
    return f.v;
  };

  stage_regs(0);
  store_lds(0);
  __syncthreads();

  int buf = 0;
  for (int k0 = 0; k0 < K; k0 += BK) {
    const bool more = (k0 + BK) < K;
    if (more) stage_regs(k0 + BK);

    v16bf fa0 = ld_frag(&sA[buf][(wm      + ml) * BK + half * 8]);
    v16bf fa1 = ld_frag(&sA[buf][(wm + 16 + ml) * BK + half * 8]);
    #pragma unroll
    for (int j = 0; j < 4; ++j) {
      v16bf fb = ld_frag(&sB[buf][(wn + j * 16 + ml) * BK + half * 8]);
      acc[0][j] = __builtin_amdgcn_wmma_f32_16x16x32_bf16(false, fa0, false, fb,
                                                          (short)0, acc[0][j], false, false);
      acc[1][j] = __builtin_amdgcn_wmma_f32_16x16x32_bf16(false, fa1, false, fb,
                                                          (short)0, acc[1][j], false, false);
    }

    if (more) store_lds(buf ^ 1);
    __syncthreads();
    buf ^= 1;
  }

  const int mbase = half * 8;
  #pragma unroll
  for (int mi = 0; mi < 2; ++mi) {
    #pragma unroll
    for (int r = 0; r < 8; ++r) {
      int gm = m0 + wm + mi * 16 + mbase + r;
      if (gm >= M) continue;
      #pragma unroll
      for (int j = 0; j < 4; ++j) {
        int gn = n0 + wn + j * 16 + ml;
        if (gn >= N) continue;
        float v = acc[mi][j][r] + bias[gn];
        out[(size_t)gm * N + gn] = v;
      }
    }
  }
}

// repack conv weight (Cout,Cin,3,3) -> (Cout, tap(9), Cin)
__global__ void k_repack_w3(const float* __restrict__ src, float* __restrict__ dst,
                            int Cin, int total) {
  int idx = blockIdx.x * 256 + threadIdx.x;
  if (idx >= total) return;
  int ci = idx % Cin;
  int t  = idx / Cin;
  int tap = t % 9;
  int co  = t / 9;
  dst[idx] = src[((size_t)co * Cin + ci) * 9 + tap];
}

// ---------------------------------------------------------------------------
// Layout change: NCHW -> rows [B*H*W, C]  (channels-last)
// ---------------------------------------------------------------------------
__global__ void k_nchw_to_rows(const float* __restrict__ src, float* __restrict__ dst,
                               int C, int HW, int total) {
  int idx = blockIdx.x * 256 + threadIdx.x;
  if (idx >= total) return;
  int c = idx % C;
  int r = idx / C;
  int b = r / HW, p = r % HW;
  dst[idx] = src[((size_t)b * C + c) * HW + p];
}

// Direct 3x3 conv, channels-last (stem conv0, Cin=3). pad=1.
__global__ void k_conv3_nhwc(const float* __restrict__ in, const float* __restrict__ w,
                             const float* __restrict__ bias, float* __restrict__ out,
                             int Hin, int Win, int Cin, int Ho, int Wo, int Cout,
                             int stride, int total) {
  int idx = blockIdx.x * 256 + threadIdx.x;
  if (idx >= total) return;
  int co = idx % Cout;
  int t  = idx / Cout;
  int wo = t % Wo; t /= Wo;
  int ho = t % Ho;
  int b  = t / Ho;
  float acc = bias[co];
  #pragma unroll
  for (int kh = 0; kh < 3; ++kh) {
    int hi = ho * stride - 1 + kh;
    if ((unsigned)hi >= (unsigned)Hin) continue;
    #pragma unroll
    for (int kw = 0; kw < 3; ++kw) {
      int wi = wo * stride - 1 + kw;
      if ((unsigned)wi >= (unsigned)Win) continue;
      const float* ip = in + (((size_t)b * Hin + hi) * Win + wi) * Cin;
      const float* wp = w + (size_t)co * Cin * 9 + kh * 3 + kw;
      for (int ci = 0; ci < Cin; ++ci) acc += ip[ci] * wp[(size_t)ci * 9];
    }
  }
  out[idx] = acc;
}

// Depthwise 3x3 conv, channels-last, float4 channel groups. pad=1. C % 4 == 0.
__global__ void k_dwconv3_nhwc4(const float* __restrict__ in, const float* __restrict__ w,
                                const float* __restrict__ bias, float* __restrict__ out,
                                int Hin, int Win, int C, int Ho, int Wo,
                                int stride, int total4) {
  int idx = blockIdx.x * 256 + threadIdx.x;
  if (idx >= total4) return;
  int C4 = C >> 2;
  int cg = idx % C4;
  int t  = idx / C4;
  int wo = t % Wo; t /= Wo;
  int ho = t % Ho;
  int b  = t / Ho;
  int c = cg * 4;
  f32x4 acc = *(const f32x4*)(bias + c);
  #pragma unroll
  for (int kh = 0; kh < 3; ++kh) {
    int hi = ho * stride - 1 + kh;
    if ((unsigned)hi >= (unsigned)Hin) continue;
    #pragma unroll
    for (int kw = 0; kw < 3; ++kw) {
      int wi = wo * stride - 1 + kw;
      if ((unsigned)wi >= (unsigned)Win) continue;
      f32x4 iv = *(const f32x4*)&in[(((size_t)b * Hin + hi) * Win + wi) * C + c];
      int tap = kh * 3 + kw;
      f32x4 wv = { w[(c + 0) * 9 + tap], w[(c + 1) * 9 + tap],
                   w[(c + 2) * 9 + tap], w[(c + 3) * 9 + tap] };
      acc += iv * wv;
    }
  }
  *(f32x4*)&out[(size_t)idx * 4] = acc;
}

// Row LayerNorm over last dim D (D % 4 == 0). g/b may be null. act==1 -> gelu.
__global__ void k_ln_rows(const float* __restrict__ in, float* __restrict__ out,
                          const float* __restrict__ g, const float* __restrict__ b,
                          int R, int D, int act) {
  int r = blockIdx.x * 256 + threadIdx.x;
  if (r >= R) return;
  const int D4 = D >> 2;
  const f32x4* xv = (const f32x4*)(in + (size_t)r * D);
  f32x4 s = {0.f, 0.f, 0.f, 0.f};
  for (int i = 0; i < D4; ++i) s += xv[i];
  float mu = (s.x + s.y + s.z + s.w) / (float)D;
  f32x4 vs = {0.f, 0.f, 0.f, 0.f};
  for (int i = 0; i < D4; ++i) { f32x4 d0 = xv[i] - mu; vs += d0 * d0; }
  float var = (vs.x + vs.y + vs.z + vs.w) / (float)D;
  float inv = rsqrtf(var + 1e-5f);
  f32x4* yv = (f32x4*)(out + (size_t)r * D);
  const f32x4* gv = (const f32x4*)g;
  const f32x4* bv = (const f32x4*)b;
  for (int i = 0; i < D4; ++i) {
    f32x4 v = (xv[i] - mu) * inv;
    if (g) v = v * gv[i] + bv[i];
    if (act == 1) v = gelu4(v);
    yv[i] = v;
  }
}

// mean over P rows per (b, c-quad): in [B,P,C] -> out [B,C]; C % 4 == 0
__global__ void k_rowmean4(const float* __restrict__ in, float* __restrict__ out,
                           int P, int C, int total4) {
  int idx = blockIdx.x * 256 + threadIdx.x;
  if (idx >= total4) return;
  int C4 = C >> 2;
  int cg = idx % C4, b = idx / C4;
  f32x4 s = {0.f, 0.f, 0.f, 0.f};
  for (int p = 0; p < P; ++p)
    s += *(const f32x4*)&in[((size_t)b * P + p) * C + cg * 4];
  *(f32x4*)&out[(size_t)b * C + cg * 4] = s * (1.f / (float)P);
}

// small FC: out[b,o] = act(sum_i in[b,i]*w[o,i]); act 1=silu, 2=sigmoid. Din%4==0
__global__ void k_fc(const float* __restrict__ in, const float* __restrict__ w,
                     float* __restrict__ out, int Din, int Dout, int act, int total) {
  int idx = blockIdx.x * 256 + threadIdx.x;
  if (idx >= total) return;
  int o = idx % Dout, b = idx / Dout;
  const f32x4* iv = (const f32x4*)(in + (size_t)b * Din);
  const f32x4* wv = (const f32x4*)(w + (size_t)o * Din);
  f32x4 s4 = {0.f, 0.f, 0.f, 0.f};
  for (int i = 0; i < (Din >> 2); ++i) s4 += iv[i] * wv[i];
  float s = s4.x + s4.y + s4.z + s4.w;
  if (act == 1) s = s / (1.f + expf(-s));
  else if (act == 2) s = 1.f / (1.f + expf(-s));
  out[idx] = s;
}

// h[b,p,c-quad] *= gate[b,c-quad]; C % 4 == 0
__global__ void k_scale_rows4(float* __restrict__ h, const float* __restrict__ gate,
                              int P, int C, int total4) {
  int idx = blockIdx.x * 256 + threadIdx.x;
  if (idx >= total4) return;
  int C4 = C >> 2;
  int cg = idx % C4;
  int b  = (idx / C4) / P;
  f32x4 gv = ((const f32x4*)(gate + (size_t)b * C))[cg];
  f32x4* hp = (f32x4*)h;
  hp[idx] *= gv;
}

__global__ void k_add4(float* __restrict__ dst, const float* __restrict__ src, int n4) {
  int i = blockIdx.x * 256 + threadIdx.x;
  if (i < n4) ((f32x4*)dst)[i] += ((const f32x4*)src)[i];
}

// window gather/scatter (float4 along d): rows [B,H,W,d] <-> windows [B,X,Y,7,7,d]
__global__ void k_win_perm4(const float* __restrict__ src, float* __restrict__ dst,
                            int H, int Wd, int d, int Xl, int Yl,
                            int grid_mode, int scatter, int total4) {
  int idx = blockIdx.x * 256 + threadIdx.x;
  if (idx >= total4) return;
  int d4 = d >> 2;
  int c4 = (idx % d4) * 4;
  int r  = idx / d4;
  int t = r;
  int wi2 = t % 49; t /= 49;
  int wi = wi2 / 7, wj = wi2 % 7;
  int gy = t % Yl; t /= Yl;
  int gx = t % Xl;
  int b  = t / Xl;
  int h, w;
  if (grid_mode) { h = wi * Xl + gx; w = wj * Yl + gy; }   // to_grid
  else           { h = gx * 7 + wi; w = gy * 7 + wj; }      // to_block
  size_t ridx = (((size_t)b * H + h) * Wd + w) * (size_t)d + c4;
  if (scatter) *(f32x4*)&dst[ridx] = *(const f32x4*)&src[(size_t)idx * 4];
  else         *(f32x4*)&dst[(size_t)idx * 4] = *(const f32x4*)&src[ridx];
}

// ---------------------------------------------------------------------------
// Fused window attention: one block per (window, head).
// n=49 tokens + 4 mem kv, dh=32, rel-pos bias. qkv layout [NW*49, 3d].
// Staging uses GLOBAL_LOAD_ASYNC_TO_LDS_B128 when available (ASYNCcnt path).
// ---------------------------------------------------------------------------
#define AN 49
#define ANM 4
#define ATOT 53
#define ADH 32

__global__ __launch_bounds__(256) void k_attn(
    const float* __restrict__ qkv, const float* __restrict__ memkv,
    const float* __restrict__ rel, float* __restrict__ out,
    int heads, int d) {
  const int blk = blockIdx.x;
  const int h = blk % heads;
  const int w = blk / heads;
  __shared__ __attribute__((aligned(16))) float sq[AN * ADH];
  __shared__ __attribute__((aligned(16))) float sk[ATOT * ADH];
  __shared__ __attribute__((aligned(16))) float sv[ATOT * ADH];
  __shared__ float ss[AN * ATOT];
  const int tid = threadIdx.x;
  const size_t rowbase = (size_t)w * AN;
  const int d3 = 3 * d;

#if defined(HAVE_ASYNC_LDS)
  for (int s = tid; s < (AN * ADH) / 4; s += 256) {
    int r = s >> 3, c4 = (s & 7) << 2;
    async_cp_b128(&qkv[(rowbase + r) * d3 + h * ADH + c4], &sq[r * ADH + c4]);
  }
  for (int s = tid; s < (ATOT * ADH) / 4; s += 256) {
    int r = s >> 3, c4 = (s & 7) << 2;
    const float* ks = (r < ANM)
        ? &memkv[((0 * heads + h) * ANM + r) * ADH + c4]
        : &qkv[(rowbase + (r - ANM)) * d3 + d + h * ADH + c4];
    const float* vs = (r < ANM)
        ? &memkv[((1 * heads + h) * ANM + r) * ADH + c4]
        : &qkv[(rowbase + (r - ANM)) * d3 + 2 * d + h * ADH + c4];
    async_cp_b128(ks, &sk[r * ADH + c4]);
    async_cp_b128(vs, &sv[r * ADH + c4]);
  }
  wait_async0();
  __syncthreads();
#else
  for (int i = tid; i < AN * ADH; i += 256) {
    int r = i >> 5, c = i & 31;
    sq[i] = qkv[(rowbase + r) * d3 + h * ADH + c];
  }
  for (int i = tid; i < ATOT * ADH; i += 256) {
    int r = i >> 5, c = i & 31;
    float kv, vv;
    if (r < ANM) {
      kv = memkv[((0 * heads + h) * ANM + r) * ADH + c];
      vv = memkv[((1 * heads + h) * ANM + r) * ADH + c];
    } else {
      kv = qkv[(rowbase + (r - ANM)) * d3 + d     + h * ADH + c];
      vv = qkv[(rowbase + (r - ANM)) * d3 + 2 * d + h * ADH + c];
    }
    sk[i] = kv; sv[i] = vv;
  }
  __syncthreads();
#endif

  for (int p = tid; p < AN * ATOT; p += 256) {
    int i = p / ATOT, j = p % ATOT;
    float acc = 0.f;
    #pragma unroll
    for (int c = 0; c < ADH; ++c) acc += sq[i * ADH + c] * sk[j * ADH + c];
    acc *= 0.17677669529663687f;   // dh^-0.5 (moved off q so staging is a raw copy)
    if (j >= ANM) {
      int jj = j - ANM;
      int xi = i / 7, yi = i % 7, xj = jj / 7, yj = jj % 7;
      int ridx = (xi - xj + 6) * 13 + (yi - yj + 6);
      acc += rel[ridx * heads + h];
    }
    ss[p] = acc;
  }
  __syncthreads();

  if (tid < AN) {
    float mx = -1e30f;
    for (int j = 0; j < ATOT; ++j) mx = fmaxf(mx, ss[tid * ATOT + j]);
    float sum = 0.f;
    for (int j = 0; j < ATOT; ++j) { float e = __expf(ss[tid * ATOT + j] - mx); ss[tid * ATOT + j] = e; sum += e; }
    float inv = 1.f / sum;
    for (int j = 0; j < ATOT; ++j) ss[tid * ATOT + j] *= inv;
  }
  __syncthreads();

  for (int p = tid; p < AN * ADH; p += 256) {
    int i = p >> 5, c = p & 31;
    float acc = 0.f;
    #pragma unroll
    for (int j = 0; j < ATOT; ++j) acc += ss[i * ATOT + j] * sv[j * ADH + c];
    out[(rowbase + i) * d + h * ADH + c] = acc;
  }
}

// ---------------------------------------------------------------------------
// Host orchestration
// ---------------------------------------------------------------------------
static inline int ceil_div_i(int a, int b) { return (a + b - 1) / b; }

extern "C" void kernel_launch(void* const* d_in, const int* in_sizes, int n_in,
                              void* d_out, int out_size, void* d_ws, size_t ws_size,
                              hipStream_t stream) {
  (void)in_sizes; (void)n_in; (void)out_size; (void)ws_size;
  const int Bn = 16;

  // ---- param cursor (setup_inputs dict insertion order) ----
  int ci = 0;
  auto nx = [&]() { return (const float*)d_in[ci++]; };
  const float* x     = nx();
  const float* st_w0 = nx(); const float* st_b0 = nx();
  const float* st_w1 = nx(); const float* st_b1 = nx();

  struct MB { const float *w1,*b1,*g1,*be1,*wd,*bd,*g2,*be2,*se1,*se2,*w2,*b2,*g3,*be3; };
  struct AT { const float *ln_g,*ln_b,*wqkv,*memkv,*rel,*wout; };
  struct FF { const float *w1,*b1,*w2,*b2; };
  MB mb[4]; AT a1[4], a2[4]; FF f1[4], f2[4];
  for (int i = 0; i < 4; ++i) {
    mb[i] = { nx(),nx(),nx(),nx(),nx(),nx(),nx(),nx(),nx(),nx(),nx(),nx(),nx(),nx() };
    a1[i] = { nx(),nx(),nx(),nx(),nx(),nx() };
    f1[i] = { nx(),nx(),nx(),nx() };
    a2[i] = { nx(),nx(),nx(),nx(),nx(),nx() };
    f2[i] = { nx(),nx(),nx(),nx() };
  }
  const float* hg  = nx(); const float* hb  = nx();
  const float* hw  = nx(); const float* hbo = nx();

  // ---- workspace arena ----
  char* wsb = (char*)d_ws;
  size_t off = 0;
  auto arena = [&](size_t bytes) -> float* {
    float* p = (float*)(wsb + off);
    off += (bytes + 255) & ~(size_t)255;
    return p;
  };
  const size_t MBy = 1024ull * 1024ull;
  float* R0   = arena(81 * MBy);    // rows ping  (max 16*112^2*96*4 = 77MB)
  float* R1   = arena(81 * MBy);    // rows pong
  float* H1   = arena(312 * MBy);   // mbconv hidden (max 16*112^2*384*4 = 308MB)
  float* H2   = arena(81 * MBy);    // mbconv hidden post-dw
  float* WIN  = arena(24 * MBy);    // window rows x6
  float* XN   = arena(24 * MBy);    // LN / proj scratch
  float* BIG  = arena(81 * MBy);    // qkv / FF hidden (max 50176*384*4 = 77MB)
  float* ATTb = arena(24 * MBy);    // attention output
  float* SM   = arena(4 * MBy);     // small vectors + repacked stem weight
  float* SM_pool = SM;
  float* SM_t1   = SM + 16384;
  float* SM_gate = SM + 32768;
  float* SM_wr   = SM + 65536;      // 96*9*96 = 82944 floats

  auto gemm = [&](const float* A, const float* Wt, const float* bias, float* C,
                  int M, int N, int K, int act) {
    dim3 g(ceil_div_i(N, BN), ceil_div_i(M, BM));
    k_gemm_bf16<<<g, 256, 0, stream>>>(A, Wt, bias, C, M, N, K, act);
  };
  auto lnrows = [&](const float* in, float* outp, const float* g, const float* b,
                    int R, int D, int act) {
    k_ln_rows<<<ceil_div_i(R, 256), 256, 0, stream>>>(in, outp, g, b, R, D, act);
  };

  // ---- stem ----
  {
    int tot = Bn * 224 * 224 * 3;
    k_nchw_to_rows<<<ceil_div_i(tot, 256), 256, 0, stream>>>(x, R0, 3, 224 * 224, tot);
    int tot0 = Bn * 112 * 112 * 96;
    k_conv3_nhwc<<<ceil_div_i(tot0, 256), 256, 0, stream>>>(
        R0, st_w0, st_b0, R1, 224, 224, 3, 112, 112, 96, 2, tot0);
    // stem conv1 (96->96 @112^2) as implicit-im2col WMMA GEMM, K = 9*96 = 864
    int trw = 96 * 9 * 96;
    k_repack_w3<<<ceil_div_i(trw, 256), 256, 0, stream>>>(st_w1, SM_wr, 96, trw);
    int M1 = Bn * 112 * 112;
    dim3 gc(ceil_div_i(96, BN), ceil_div_i(M1, BM));
    k_conv_gemm_bf16<<<gc, 256, 0, stream>>>(
        R1, SM_wr, st_b1, R0, 112, 112, 96, 112, 112, 1, M1, 96, 864);
  }
  float* cur = R0; float* alt = R1;
  int H = 112;

  struct BlkMeta { int din, dout, down; };
  BlkMeta bm[4] = { {96,96,1}, {96,96,0}, {96,192,1}, {192,192,0} };

  for (int bi = 0; bi < 4; ++bi) {
    const int din = bm[bi].din, dout = bm[bi].dout, down = bm[bi].down;
    const int hid = 4 * dout, shr = hid / 4;

    // ---- MBConv ----
    {
      int Pin = H * H;
      gemm(cur, mb[bi].w1, mb[bi].b1, H1, Bn * Pin, hid, din, 0);
      lnrows(H1, H1, mb[bi].g1, mb[bi].be1, Bn * Pin, hid, 1);
      int Ho = down ? H / 2 : H;
      int Po = Ho * Ho;
      int totd = Bn * Po * hid;
      k_dwconv3_nhwc4<<<ceil_div_i(totd / 4, 256), 256, 0, stream>>>(
          H1, mb[bi].wd, mb[bi].bd, H2, H, H, hid, Ho, Ho, down ? 2 : 1, totd / 4);
      lnrows(H2, H2, mb[bi].g2, mb[bi].be2, Bn * Po, hid, 1);
      // squeeze-excitation
      int tp = Bn * hid;
      k_rowmean4<<<ceil_div_i(tp / 4, 256), 256, 0, stream>>>(H2, SM_pool, Po, hid, tp / 4);
      int t1 = Bn * shr;
      k_fc<<<ceil_div_i(t1, 256), 256, 0, stream>>>(SM_pool, mb[bi].se1, SM_t1, hid, shr, 1, t1);
      k_fc<<<ceil_div_i(tp, 256), 256, 0, stream>>>(SM_t1, mb[bi].se2, SM_gate, shr, hid, 2, tp);
      k_scale_rows4<<<ceil_div_i(totd / 4, 256), 256, 0, stream>>>(H2, SM_gate, Po, hid, totd / 4);
      gemm(H2, mb[bi].w2, mb[bi].b2, alt, Bn * Po, dout, hid, 0);
      lnrows(alt, alt, mb[bi].g3, mb[bi].be3, Bn * Po, dout, 0);
      if (!down) { // residual (din==dout exactly when !down in this config)
        int ta = Bn * Po * dout;
        k_add4<<<ceil_div_i(ta / 4, 256), 256, 0, stream>>>(alt, cur, ta / 4);
      }
      float* t = cur; cur = alt; alt = t;
      H = Ho;
    }

    // ---- block attention + FF, then grid attention + FF ----
    for (int mode = 0; mode < 2; ++mode) {
      const AT& a = (mode == 0) ? a1[bi] : a2[bi];
      const FF& f = (mode == 0) ? f1[bi] : f2[bi];
      const int d = dout, heads = d / 32, Xl = H / 7;
      const int NW = Bn * Xl * Xl, R = NW * 49;
      const int tot = R * d;

      k_win_perm4<<<ceil_div_i(tot / 4, 256), 256, 0, stream>>>(
          cur, WIN, H, H, d, Xl, Xl, mode, 0, tot / 4);
      // attention
      lnrows(WIN, XN, a.ln_g, a.ln_b, R, d, 0);
      gemm(XN, a.wqkv, nullptr, BIG, R, 3 * d, d, 0);
      k_attn<<<NW * heads, 256, 0, stream>>>(BIG, a.memkv, a.rel, ATTb, heads, d);
      gemm(ATTb, a.wout, nullptr, XN, R, d, d, 0);
      k_add4<<<ceil_div_i(tot / 4, 256), 256, 0, stream>>>(WIN, XN, tot / 4);
      // feed-forward
      lnrows(WIN, XN, nullptr, nullptr, R, d, 0);
      gemm(XN, f.w1, f.b1, BIG, R, 4 * d, d, 1);
      gemm(BIG, f.w2, f.b2, XN, R, d, 4 * d, 0);
      k_add4<<<ceil_div_i(tot / 4, 256), 256, 0, stream>>>(WIN, XN, tot / 4);

      k_win_perm4<<<ceil_div_i(tot / 4, 256), 256, 0, stream>>>(
          WIN, alt, H, H, d, Xl, Xl, mode, 1, tot / 4);
      float* t = cur; cur = alt; alt = t;
    }
  }

  // ---- head ----
  {
    const int ed = 192;
    int tp = Bn * ed;
    k_rowmean4<<<ceil_div_i(tp / 4, 256), 256, 0, stream>>>(cur, SM_pool, H * H, ed, tp / 4);
    lnrows(SM_pool, SM_t1, hg, hb, Bn, ed, 0);
    gemm(SM_t1, hw, hbo, (float*)d_out, Bn, 1000, ed, 0);
  }
}
